// MultiHeadAttnBlock_44212393345067
// MI455X (gfx1250) — matrix operand order repaired and checked by reading, not compiled, not run on hardware
//
#include <hip/hip_runtime.h>
#include <cstdint>
#include <cstddef>

typedef __attribute__((ext_vector_type(8)))  unsigned short u16x8;
typedef __attribute__((ext_vector_type(16))) unsigned short u16x16;
typedef __attribute__((ext_vector_type(16))) __bf16         v16bf;
typedef __attribute__((ext_vector_type(8)))  float          v8f;

#define DEVINL __device__ __forceinline__

static constexpr float EPS_   = 1e-5f;
static constexpr float SCALE_ = 0.35355339059327373f; // 64^-0.25

DEVINL unsigned short f2bf(float f) {
  unsigned u = __builtin_bit_cast(unsigned, f);
  u += 0x7fffu + ((u >> 16) & 1u);               // round-to-nearest-even
  return (unsigned short)(u >> 16);
}

// low 32 bits of a generic pointer to __shared__ = LDS byte offset
DEVINL unsigned ldsoff(const void* p) { return (unsigned)(uintptr_t)p; }

// CDNA5 async global->LDS copy, 16 bytes per lane (ASYNCcnt-tracked)
DEVINL void async_ld128(unsigned lds_byte_off, const void* gaddr) {
  asm volatile("global_load_async_to_lds_b128 %0, %1, off"
               :: "v"(lds_byte_off), "v"(gaddr) : "memory");
}
DEVINL void wait_async() { asm volatile("s_wait_asynccnt 0x0" ::: "memory"); }

// 16 contiguous bf16 halves -> B-fragment (B 32x16: lane n = l&15, k = (l>>4)*16 + e)
DEVINL v16bf ld16(const unsigned short* p) {
  u16x8 lo = *(const u16x8*)p;
  u16x8 hi = *(const u16x8*)(p + 8);
  return __builtin_bit_cast(v16bf,
      __builtin_shufflevector(lo, hi, 0,1,2,3,4,5,6,7,8,9,10,11,12,13,14,15));
}

// A-fragment from a 32-half (K=32) row window: k(e) = sel*16 + e
DEVINL v16bf ld_afrag(const unsigned short* win, int sel) {
  u16x8 lo = *(const u16x8*)(win + sel * 16);
  u16x8 hi = *(const u16x8*)(win + sel * 16 + 8);
  return __builtin_bit_cast(v16bf,
      __builtin_shufflevector(lo, hi, 0,1,2,3,4,5,6,7,8,9,10,11,12,13,14,15));
}

DEVINL v8f wmma_bf16(v16bf a, v16bf b, v8f c) {
  return __builtin_amdgcn_wmma_f32_16x16x32_bf16(false, a, false, b, (short)0, c,
                                                 false, false);
}

// ---------------- GroupNorm ----------------
__global__ __launch_bounds__(256) void gn_stats_k(const float* __restrict__ x,
                                                  float* __restrict__ stats) {
  __shared__ float rs[256], rq[256];
  const int tid = threadIdx.x;
  const size_t base = (size_t)blockIdx.x * 32768u;   // group = contiguous 16ch*2048
  float s = 0.f, q = 0.f;
  for (int i = tid; i < 32768; i += 256) {
    float v = x[base + i];
    s += v; q += v * v;
  }
  rs[tid] = s; rq[tid] = q;
  __syncthreads();
  for (int off = 128; off > 0; off >>= 1) {
    if (tid < off) { rs[tid] += rs[tid + off]; rq[tid] += rq[tid + off]; }
    __syncthreads();
  }
  if (tid == 0) {
    float mu  = rs[0] * (1.f / 32768.f);
    float var = rq[0] * (1.f / 32768.f) - mu * mu;
    stats[blockIdx.x * 2 + 0] = mu;
    stats[blockIdx.x * 2 + 1] = rsqrtf(var + EPS_);
  }
}

__global__ __launch_bounds__(256) void gn_apply_k(const float* __restrict__ x,
                                                  const float* __restrict__ gw,
                                                  const float* __restrict__ gb,
                                                  const float* __restrict__ stats,
                                                  unsigned short* __restrict__ ht) {
  const int idx = blockIdx.x * 256 + threadIdx.x;    // over 8*512*2048
  const int b   = idx >> 20;
  const int rem = idx & ((1 << 20) - 1);
  const int c   = rem >> 11;
  const int t   = rem & 2047;
  const int sg  = (b << 5) | (c >> 4);
  const float mu = stats[sg * 2], rstd = stats[sg * 2 + 1];
  const float v  = (x[idx] - mu) * rstd * gw[c] + gb[c];
  ht[((size_t)(b * 2048 + t)) * 512 + c] = f2bf(v);  // transpose to [b][t][c]
}

__global__ __launch_bounds__(256) void cvt_bf16_k(const float* __restrict__ in,
                                                  unsigned short* __restrict__ out,
                                                  int n) {
  int i = blockIdx.x * 256 + threadIdx.x;
  if (i < n) out[i] = f2bf(in[i]);
}

// ---------------- 64x64 GEMM core, async double-buffered ----------------
// Wg: row-major [64 rows of M][K]   (ptr at m0,0)
// Xg: row-major [64 rows of N][K]   (ptr at n0,0)  (B stored transposed)
// As/Bs: 2 x 64x32 halves each
DEVINL void gemm_issue(const unsigned short* Wg, const unsigned short* Xg,
                       size_t goff, int k0, unsigned ab, unsigned bb) {
  async_ld128(ab,      Wg + goff + k0);
  async_ld128(ab + 16, Wg + goff + k0 + 8);
  async_ld128(bb,      Xg + goff + k0);
  async_ld128(bb + 16, Xg + goff + k0 + 8);
}

DEVINL void gemm64x64(const unsigned short* __restrict__ Wg,
                      const unsigned short* __restrict__ Xg,
                      int K, unsigned short* As, unsigned short* Bs, v8f acc[4]) {
  const int tid  = threadIdx.x;
  const int lane = tid & 31, wave = tid >> 5;
  const int row  = tid >> 1, seg = tid & 1;
  const v8f z = {0.f, 0.f, 0.f, 0.f, 0.f, 0.f, 0.f, 0.f};
  acc[0] = z; acc[1] = z; acc[2] = z; acc[3] = z;
  const size_t goff = (size_t)row * K + seg * 16;
  const unsigned aoff = (unsigned)(row * 64 + seg * 32);     // byte offset in tile
  const unsigned ab[2] = { ldsoff(As) + aoff, ldsoff(As) + 4096u + aoff };
  const unsigned bb[2] = { ldsoff(Bs) + aoff, ldsoff(Bs) + 4096u + aoff };

  gemm_issue(Wg, Xg, goff, 0, ab[0], bb[0]);
  for (int k0 = 0; k0 < K; k0 += 32) {
    const int buf = (k0 >> 5) & 1;
    wait_async();                       // this wave's tile loads done
    __syncthreads();                    // everyone's loads done; prev buffer free
    if (k0 + 32 < K)
      gemm_issue(Wg, Xg, goff, k0 + 32, ab[buf ^ 1], bb[buf ^ 1]);
    const unsigned short* Asb = As + buf * 2048;
    const unsigned short* Bsb = Bs + buf * 2048;
    // preload all fragments into distinct regs -> single dscnt wait, dense WMMA issue
    const v16bf af  = ld_afrag(Asb + (wave * 16 + (lane & 15)) * 32, lane >> 4);
    const int   bo  = (lane & 15) * 32 + ((lane >> 4) << 4);
    const v16bf bf0 = ld16(Bsb + bo);
    const v16bf bf1 = ld16(Bsb + 16 * 32 + bo);
    const v16bf bf2 = ld16(Bsb + 32 * 32 + bo);
    const v16bf bf3 = ld16(Bsb + 48 * 32 + bo);
    acc[0] = wmma_bf16(af, bf0, acc[0]);
    acc[1] = wmma_bf16(af, bf1, acc[1]);
    acc[2] = wmma_bf16(af, bf2, acc[2]);
    acc[3] = wmma_bf16(af, bf3, acc[3]);
  }
}

// ---------------- QKV projection ----------------
__global__ __launch_bounds__(128) void qkv_gemm_k(const unsigned short* __restrict__ Wb,
                                                  const unsigned short* __restrict__ ht,
                                                  const float* __restrict__ qkv_b,
                                                  unsigned short* __restrict__ qt,
                                                  unsigned short* __restrict__ kt,
                                                  unsigned short* __restrict__ vm) {
  __shared__ __attribute__((aligned(16))) unsigned short As[2 * 64 * 32];
  __shared__ __attribute__((aligned(16))) unsigned short Bs[2 * 64 * 32];
  const int o0 = blockIdx.x * 64, t0 = blockIdx.y * 64, b = blockIdx.z;
  v8f acc[4];
  gemm64x64(Wb + (size_t)o0 * 512, ht + ((size_t)b * 2048 + t0) * 512, 512, As, Bs, acc);
  const int lane = threadIdx.x & 31, wave = threadIdx.x >> 5;
  const int mrow = wave * 16 + ((lane >> 4) << 3);
#pragma unroll
  for (int j = 0; j < 4; ++j) {
    const int t = t0 + j * 16 + (lane & 15);
#pragma unroll
    for (int r = 0; r < 8; ++r) {
      const int o = o0 + mrow + r;
      float v = acc[j][r] + qkv_b[o];
      const int which = o >> 9;            // 0=q,1=k,2=v
      const int oc = o & 511;
      const int head = b * 8 + (oc >> 6);
      const int cc = oc & 63;
      if (which == 0)      qt[((size_t)head * 2048 + t) * 64 + cc] = f2bf(v * SCALE_);
      else if (which == 1) kt[((size_t)head * 2048 + t) * 64 + cc] = f2bf(v * SCALE_);
      else                 vm[((size_t)head * 64 + cc) * 2048 + t] = f2bf(v);
    }
  }
}

// ---------------- Flash attention (per head, 64-query tiles) ----------------
DEVINL void attn_issue(const unsigned short* kt, const unsigned short* vm,
                       int bh, int s0, int tid, unsigned kbase, unsigned vbase) {
  // K tile: [s][c] -> contiguous 8KB; thread copies 4 x 16B units
  const unsigned short* ksrc = kt + ((size_t)bh * 2048 + s0) * 64;
#pragma unroll
  for (int i = 0; i < 4; ++i) {
    const int u = i * 128 + tid;
    async_ld128(kbase + (unsigned)u * 16u, ksrc + (size_t)u * 8);
  }
  // V tile: [c][s] 64 rows of 64 halves; thread = (row, half-row segment)
  const int vrow = tid >> 1, vseg = tid & 1;
  const unsigned short* vsrc = vm + ((size_t)bh * 64 + vrow) * 2048 + s0 + vseg * 32;
  const unsigned vdst = vbase + (unsigned)(vrow * 128 + vseg * 64);
#pragma unroll
  for (int i = 0; i < 4; ++i) async_ld128(vdst + i * 16, vsrc + i * 8);
}

__global__ __launch_bounds__(128) void attn_k(const unsigned short* __restrict__ qt,
                                              const unsigned short* __restrict__ kt,
                                              const unsigned short* __restrict__ vm,
                                              unsigned short* __restrict__ at) {
  __shared__ __attribute__((aligned(16))) unsigned short Ks[2 * 64 * 64];
  __shared__ __attribute__((aligned(16))) unsigned short Vs[2 * 64 * 64];
  __shared__ __attribute__((aligned(16))) unsigned short Ps[4 * 16 * 64];
  const int tid = threadIdx.x, lane = tid & 31, wave = tid >> 5;
  const int bh = blockIdx.y, q0 = blockIdx.x * 64;
  const int b = bh >> 3, h = bh & 7;

  // Q fragments (A-layout, K=c split into two 32-chunks) straight from global
  const unsigned short* qrow =
      qt + ((size_t)bh * 2048 + q0 + wave * 16 + (lane & 15)) * 64;
  const v16bf qf0 = ld_afrag(qrow, lane >> 4);
  const v16bf qf1 = ld_afrag(qrow + 32, lane >> 4);

  const v8f z = {0.f, 0.f, 0.f, 0.f, 0.f, 0.f, 0.f, 0.f};
  v8f accO[4] = {z, z, z, z};
  float mi[8], li[8];
#pragma unroll
  for (int r = 0; r < 8; ++r) { mi[r] = -1e30f; li[r] = 0.f; }

  const unsigned kb[2] = { ldsoff(Ks), ldsoff(Ks) + 8192u };
  const unsigned vb[2] = { ldsoff(Vs), ldsoff(Vs) + 8192u };

  attn_issue(kt, vm, bh, 0, tid, kb[0], vb[0]);
  for (int s0 = 0; s0 < 2048; s0 += 64) {
    const int buf = (s0 >> 6) & 1;
    wait_async();
    __syncthreads();
    if (s0 + 64 < 2048)
      attn_issue(kt, vm, bh, s0 + 64, tid, kb[buf ^ 1], vb[buf ^ 1]);
    const unsigned short* Ksb = Ks + buf * 4096;
    const unsigned short* Vsb = Vs + buf * 4096;
    const int fo = (lane & 15) * 64 + ((lane >> 4) << 4);

    // S = Q^T K  (16 q-rows per wave x 64 keys); preload frags, dense WMMA issue
    v8f sA[4] = {z, z, z, z};
#pragma unroll
    for (int kk = 0; kk < 2; ++kk) {
      const v16bf a   = kk ? qf1 : qf0;
      const v16bf kf0 = ld16(Ksb + fo + kk * 32);
      const v16bf kf1 = ld16(Ksb + 16 * 64 + fo + kk * 32);
      const v16bf kf2 = ld16(Ksb + 32 * 64 + fo + kk * 32);
      const v16bf kf3 = ld16(Ksb + 48 * 64 + fo + kk * 32);
      sA[0] = wmma_bf16(a, kf0, sA[0]);
      sA[1] = wmma_bf16(a, kf1, sA[1]);
      sA[2] = wmma_bf16(a, kf2, sA[2]);
      sA[3] = wmma_bf16(a, kf3, sA[3]);
    }

    // Online softmax (rows live across 16-lane halves; xor 1/2/4/8 stays in-half)
#pragma unroll
    for (int r = 0; r < 8; ++r) {
      float mx = fmaxf(fmaxf(sA[0][r], sA[1][r]), fmaxf(sA[2][r], sA[3][r]));
      mx = fmaxf(mx, __shfl_xor(mx, 1, 32));
      mx = fmaxf(mx, __shfl_xor(mx, 2, 32));
      mx = fmaxf(mx, __shfl_xor(mx, 4, 32));
      mx = fmaxf(mx, __shfl_xor(mx, 8, 32));
      const float mnew  = fmaxf(mi[r], mx);
      const float alpha = __expf(mi[r] - mnew);
      float rs = 0.f;
#pragma unroll
      for (int j = 0; j < 4; ++j) {
        float p = __expf(sA[j][r] - mnew);
        sA[j][r] = p; rs += p;
      }
      rs += __shfl_xor(rs, 1, 32);
      rs += __shfl_xor(rs, 2, 32);
      rs += __shfl_xor(rs, 4, 32);
      rs += __shfl_xor(rs, 8, 32);
      li[r] = li[r] * alpha + rs;
      mi[r] = mnew;
#pragma unroll
      for (int j = 0; j < 4; ++j) accO[j][r] *= alpha;
    }

    // C-layout P -> wave-private LDS in A-friendly [m][s] layout
    unsigned short* pw = Ps + wave * 1024;
#pragma unroll
    for (int j = 0; j < 4; ++j)
#pragma unroll
      for (int r = 0; r < 8; ++r) {
        const int m = r + ((lane >> 4) << 3);
        pw[m * 64 + j * 16 + (lane & 15)] = f2bf(sA[j][r]);
      }
    asm volatile("s_wait_dscnt 0" ::: "memory");

    // O += P * V ; preload frags, dense WMMA issue
#pragma unroll
    for (int kk = 0; kk < 2; ++kk) {
      const v16bf pf  = ld_afrag(pw + (lane & 15) * 64 + kk * 32, lane >> 4);
      const v16bf vf0 = ld16(Vsb + fo + kk * 32);
      const v16bf vf1 = ld16(Vsb + 16 * 64 + fo + kk * 32);
      const v16bf vf2 = ld16(Vsb + 32 * 64 + fo + kk * 32);
      const v16bf vf3 = ld16(Vsb + 48 * 64 + fo + kk * 32);
      accO[0] = wmma_bf16(pf, vf0, accO[0]);
      accO[1] = wmma_bf16(pf, vf1, accO[1]);
      accO[2] = wmma_bf16(pf, vf2, accO[2]);
      accO[3] = wmma_bf16(pf, vf3, accO[3]);
    }
  }

  // finalize: divide by row sums, write a^T in [b][t][c] (K-contig for proj GEMM)
#pragma unroll
  for (int r = 0; r < 8; ++r) {
    const float inv = 1.f / li[r];
    const int t = q0 + wave * 16 + r + ((lane >> 4) << 3);
#pragma unroll
    for (int j = 0; j < 4; ++j) {
      const int c = h * 64 + j * 16 + (lane & 15);
      at[((size_t)(b * 2048 + t)) * 512 + c] = f2bf(accO[j][r] * inv);
    }
  }
}

// ---------------- Output projection + residual ----------------
__global__ __launch_bounds__(128) void proj_gemm_k(const unsigned short* __restrict__ Wb,
                                                   const unsigned short* __restrict__ at,
                                                   const float* __restrict__ proj_b,
                                                   const float* __restrict__ x,
                                                   float* __restrict__ out) {
  __shared__ __attribute__((aligned(16))) unsigned short As[2 * 64 * 32];
  __shared__ __attribute__((aligned(16))) unsigned short Bs[2 * 64 * 32];
  const int o0 = blockIdx.x * 64, t0 = blockIdx.y * 64, b = blockIdx.z;
  v8f acc[4];
  gemm64x64(Wb + (size_t)o0 * 512, at + ((size_t)b * 2048 + t0) * 512, 512, As, Bs, acc);
  const int lane = threadIdx.x & 31, wave = threadIdx.x >> 5;
  const int mrow = wave * 16 + ((lane >> 4) << 3);
#pragma unroll
  for (int j = 0; j < 4; ++j) {
    const int t = t0 + j * 16 + (lane & 15);
#pragma unroll
    for (int r = 0; r < 8; ++r) {
      const int o = o0 + mrow + r;
      const size_t idx = ((size_t)(b * 512 + o)) * 2048 + t;
      out[idx] = x[idx] + acc[j][r] + proj_b[o];
    }
  }
}

extern "C" void kernel_launch(void* const* d_in, const int* in_sizes, int n_in,
                              void* d_out, int out_size, void* d_ws, size_t ws_size,
                              hipStream_t stream) {
  (void)in_sizes; (void)n_in; (void)out_size; (void)ws_size;
  const float* x      = (const float*)d_in[0];
  const float* gn_w   = (const float*)d_in[1];
  const float* gn_b   = (const float*)d_in[2];
  const float* qkv_w  = (const float*)d_in[3];
  const float* qkv_b  = (const float*)d_in[4];
  const float* proj_w = (const float*)d_in[5];
  const float* proj_b = (const float*)d_in[6];
  float* out = (float*)d_out;

  char* ws = (char*)d_ws;
  constexpr size_t E = 8ull * 2048 * 512;           // elements per bf16 tensor
  float*          stats = (float*)ws;               // 256 * 2 f32
  unsigned short* ht  = (unsigned short*)(ws + 4096);
  unsigned short* qt  = ht + E;
  unsigned short* kt  = qt + E;
  unsigned short* vm  = kt + E;
  unsigned short* at  = vm + E;
  unsigned short* qwb = at + E;                     // 1536*512 bf16
  unsigned short* pwb = qwb + 1536ull * 512;        // 512*512 bf16

  gn_stats_k<<<256, 256, 0, stream>>>(x, stats);
  gn_apply_k<<<(8 * 512 * 2048) / 256, 256, 0, stream>>>(x, gn_w, gn_b, stats, ht);
  cvt_bf16_k<<<(1536 * 512) / 256, 256, 0, stream>>>(qkv_w, qwb, 1536 * 512);
  cvt_bf16_k<<<(512 * 512) / 256, 256, 0, stream>>>(proj_w, pwb, 512 * 512);
  qkv_gemm_k<<<dim3(24, 32, 8), 128, 0, stream>>>(qwb, ht, qkv_b, qt, kt, vm);
  attn_k<<<dim3(32, 64), 128, 0, stream>>>(qt, kt, vm, at);
  proj_gemm_k<<<dim3(8, 32, 8), 128, 0, stream>>>(pwb, at, proj_b, x, out);
}
